// GHNN_4114578670329
// MI455X (gfx1250) — matrix operand order repaired and checked by reading, not compiled, not run on hardware
//
#include <hip/hip_runtime.h>
#include <hip/hip_bf16.h>

typedef __attribute__((ext_vector_type(16))) _Float16 v16h;
typedef __attribute__((ext_vector_type(8)))  _Float16 v8h;
typedef __attribute__((ext_vector_type(8)))  float    v8f;
typedef __attribute__((ext_vector_type(4)))  unsigned int v4u;
typedef __attribute__((ext_vector_type(8)))  int v8i;
typedef __attribute__((ext_vector_type(4)))  int v4i;

#define USE_TDM 1   // Tensor Data Mover staging of B tiles (set 0 -> plain LDS staging)

#define BATCH 2048
#define LSEQ  10
#define HMAX  10
#define NE    20000
#define NR    200
#define DIM   200

// Padded GEMM dims (K mult of 32, N mult of 64, M mult of 128)
#define KP_V  800     // 4*DIM gates K (25*32)
#define NP_G  1408    // 7*DIM=1400 -> 22*64
#define KP_D  224     // DIM -> 7*32
#define NP_D  256     // DIM -> 4*64
#define KP_F  608     // 3*DIM=600 -> 19*32
#define NP_E  20032   // NE -> 313*64

#define CKMAX 224     // K-chunk staged in LDS: 64 rows * 224 halfs = 28.7 KB

__device__ __forceinline__ float sigmoidf_(float x) { return 1.0f / (1.0f + __expf(-x)); }
__device__ __forceinline__ float softplus_b_(float x) {
  float z = 10.0f * x;
  float sp = (z > 15.0f) ? z : log1pf(__expf(z));
  return 0.1f * sp;
}

#if USE_TDM
// Issue one TDM 2D tile load: kc x 64 rows, global row stride ldg elements,
// packed contiguously at LDS byte offset ldsdst. (ISA ch.8 descriptor layout.)
__device__ __forceinline__ void tdm_issue_b_tile(const _Float16* gsrc, int kc, int ldg,
                                                 unsigned ldsdst)
{
  unsigned long long ga = (unsigned long long)(size_t)gsrc;
  v4u g0;
  g0.x = 1u;                                   // count=1, user mode
  g0.y = ldsdst;                               // lds_addr (bytes)
  g0.z = (unsigned)(ga & 0xFFFFFFFFu);         // global_addr[31:0]
  g0.w = (unsigned)(ga >> 32) | (2u << 30);    // global_addr[56:32] | type=2
  v8i g1;
  g1[0] = (1 << 16);                           // data_size=1 -> 2-byte elements
  g1[1] = (int)((unsigned)kc << 16);           // tensor_dim0[15:0]=kc
  g1[2] = (64 << 16);                          // tensor_dim0[31:16]=0 | tensor_dim1=64
  g1[3] = (int)((unsigned)kc << 16);           // tensor_dim1[31:16]=0 | tile_dim0=kc
  g1[4] = 64;                                  // tile_dim1=64, tile_dim2=0 (unused)
  g1[5] = ldg;                                 // tensor_dim0_stride[31:0] (elements)
  g1[6] = 0;                                   // stride[47:32]=0, dim1_stride=0
  g1[7] = 0;
  v4i gz4 = {0, 0, 0, 0};                      // groups 2/3: dims unused (2D tensor)
  v8i gz8 = {0, 0, 0, 0, 0, 0, 0, 0};          // extra group (clang-23 6-arg form)
  __builtin_amdgcn_tensor_load_to_lds(g0, g1, gz4, gz4, gz8, 0);
}
#endif

// ---------------------------------------------------------------------------
// Block-tiled WMMA GEMM:  C[M,Nstore] = A_f16[M,KP] * Bt_f16[Npad,KP]^T
// Block = 8 waves = 128 rows of M x 64 cols of N. The 64-row B tile is staged
// into double-buffered LDS by the TDM (chunk c+1 DMA overlaps chunk c math);
// KP is a template constant so chunk/k loops fully unroll and the scheduler
// can pipeline ds_load ahead of v_wmma. Each wave: 16x64 tile, 4 accumulators.
// mode 0: f32 store (+bias)   mode 1: f32 softplus_b store   mode 2: f16 store
// ---------------------------------------------------------------------------
template <int KP>
__global__ void ghnn_wmma_gemm(const _Float16* __restrict__ A,
                               const _Float16* __restrict__ Bt,
                               float* __restrict__ C,
                               _Float16* __restrict__ Ch,
                               const float* __restrict__ bias,
                               int Nstore, int ldc, int ldch, int mode)
{
  constexpr int NCHUNK = (KP + CKMAX - 1) / CKMAX;
  __shared__ _Float16 ldsB[2][64 * CKMAX];

  const int tid  = threadIdx.x;
  const int wv   = tid >> 5;
  const int lane = tid & 31;
  const int n0   = blockIdx.x << 6;
  const int m0   = (blockIdx.y << 7) + (wv << 4);
  const int rsel = lane & 15;
  const int hi   = lane >> 4;

  // A 16-bit fragment: lane<16 holds K [0..15]; lane>=16 [16..31]
  const _Float16* ap   = A + (size_t)(m0 + rsel) * KP + hi * 16;
  const _Float16* brow = Bt + (size_t)n0 * KP;

#if USE_TDM
  unsigned ldsoff[2];
  ldsoff[0] = (unsigned)(size_t)(__attribute__((address_space(3))) _Float16*)&ldsB[0][0];
  ldsoff[1] = (unsigned)(size_t)(__attribute__((address_space(3))) _Float16*)&ldsB[1][0];
  if (wv == 0)   // prologue: DMA chunk 0
    tdm_issue_b_tile(brow, (KP < CKMAX ? KP : CKMAX), KP, ldsoff[0]);
#endif

  v8f accs[4] = {};
#pragma unroll
  for (int c = 0; c < NCHUNK; ++c) {
    const int kb = c * CKMAX;
    const int kc = (KP - kb) < CKMAX ? (KP - kb) : CKMAX;   // compile-time after unroll

    __syncthreads();   // all waves done reading buf[(c+1)&1] (chunk c-1's data)
#if USE_TDM
    if (wv == 0) {
      if (c + 1 < NCHUNK) {
        const int kb2 = (c + 1) * CKMAX;
        const int kc2 = (KP - kb2) < CKMAX ? (KP - kb2) : CKMAX;
        tdm_issue_b_tile(brow + kb2, kc2, KP, ldsoff[(c + 1) & 1]);
        __builtin_amdgcn_s_wait_tensorcnt(1);   // in-order TDM: chunk c has landed
      } else {
        __builtin_amdgcn_s_wait_tensorcnt(0);
      }
    }
#else
    {
      const _Float16* bsrc = brow + kb;
      const int chunks = (64 * kc) >> 3;          // v8h chunks, kc mult of 32
      for (int i = tid; i < chunks; i += 256) {
        int off2 = i << 3;
        int row = off2 / kc;
        int col = off2 - row * kc;
        *(v8h*)(&ldsB[c & 1][0] + off2) = *(const v8h*)(bsrc + (size_t)row * KP + col);
      }
    }
#endif
    __syncthreads();   // chunk c visible to all waves

    const _Float16* lbb = &ldsB[c & 1][0];
#pragma unroll
    for (int k = 0; k < kc; k += 32) {
      v8h alo = *(const v8h*)(ap + kb + k);
      v8h ahi = *(const v8h*)(ap + kb + k + 8);
      v16h a = __builtin_shufflevector(alo, ahi,
                                       0,1,2,3,4,5,6,7,8,9,10,11,12,13,14,15);
#pragma unroll
      for (int j = 0; j < 4; ++j) {
        // B fragment from LDS: lane's column = j*16 + (lane&15), K contiguous 16
        const _Float16* lb = lbb + (j * 16 + rsel) * kc + hi * 16 + k;
        v16h b = *(const v16h*)lb;
        accs[j] = __builtin_amdgcn_wmma_f32_16x16x32_f16(
            false, a, false, b, (short)0, accs[j], false, false);
      }
    }
  }

  // C/D layout: lane holds col n0+16j+(lane&15); VGPR r -> row m0 + 8*(lane>>4) + r
  const int mb = m0 + (hi << 3);
#pragma unroll
  for (int j = 0; j < 4; ++j) {
    int n = n0 + (j << 4) + rsel;
    if (n >= Nstore) continue;
    float badd = bias ? bias[n] : 0.0f;
#pragma unroll
    for (int r2 = 0; r2 < 8; ++r2) {
      float x = accs[j][r2] + badd;
      int row = mb + r2;
      if (mode == 1) x = softplus_b_(x);
      if (mode == 2) Ch[(size_t)row * ldch + n] = (_Float16)x;
      else           C [(size_t)row * ldc  + n] = x;
    }
  }
}

// ---------------------------------------------------------------------------
// Masked-mean neighbor aggregation: agg[b,l,:] = mean(ent[hist_events[b,l,0:len]])
// ---------------------------------------------------------------------------
__global__ void ghnn_aggregate(const int* __restrict__ hev, const int* __restrict__ hlen,
                               const float* __restrict__ ent, float* __restrict__ agg)
{
  const int row = blockIdx.x;          // b*LSEQ + l
  const int d   = threadIdx.x;
  if (d >= DIM) return;
  int len = hlen[row];
  if (len > HMAX) len = HMAX;
  float s = 0.0f;
  for (int h = 0; h < len; ++h) {
    int e = hev[(size_t)row * HMAX + h];
    s += ent[(size_t)e * DIM + d];     // 16MB table -> L2 resident
  }
  int dn = len > 0 ? len : 1;
  agg[(size_t)row * DIM + d] = s / (float)dn;
}

// ---------------------------------------------------------------------------
// Weight transpose+convert: Wt[Npad,Kp] (f16) = W[Kin,Nin]^T, zero padded
// ---------------------------------------------------------------------------
__global__ void ghnn_prep_wT(const float* __restrict__ W, _Float16* __restrict__ Wt,
                             int Kin, int Nin, int Npad, int Kp)
{
  int idx = blockIdx.x * blockDim.x + threadIdx.x;
  if (idx >= Npad * Kp) return;
  int n = idx / Kp, k = idx - n * Kp;
  float v = (n < Nin && k < Kin) ? W[(size_t)k * Nin + n] : 0.0f;
  Wt[idx] = (_Float16)v;
}

// ent_embeds [NE,DIM] f32 -> e16 [NP_E,KP_D] f16 zero-padded (B operand of final GEMM)
__global__ void ghnn_conv_ent(const float* __restrict__ ent, _Float16* __restrict__ e16)
{
  int idx = blockIdx.x * blockDim.x + threadIdx.x;
  if (idx >= NP_E * KP_D) return;
  int n = idx / KP_D, k = idx - n * KP_D;
  float v = (n < NE && k < DIM) ? ent[(size_t)n * DIM + k] : 0.0f;
  e16[idx] = (_Float16)v;
}

// ---------------------------------------------------------------------------
// Per-call state init: x_static halves of v, feat static cols, zero states/pads
// ---------------------------------------------------------------------------
__global__ void ghnn_setup(const int* __restrict__ s, const int* __restrict__ r,
                           const float* __restrict__ ent, const float* __restrict__ rel,
                           const float* __restrict__ agg,
                           _Float16* __restrict__ v16, _Float16* __restrict__ h16,
                           float* __restrict__ c, float* __restrict__ ctgt,
                           _Float16* __restrict__ feat16, _Float16* __restrict__ int16_)
{
  const int b = blockIdx.x;
  const int d = threadIdx.x;
  const int si = s[b], ri = r[b];
  if (d < DIM) {
    float se = ent[(size_t)si * DIM + d];
    float re = rel[(size_t)ri * DIM + d];
    _Float16* vb = v16 + (size_t)b * KP_V;
    vb[d]            = (_Float16)se;                              // x_static: ent
    vb[DIM + d]      = (_Float16)re;                              // x_static: rel
    vb[2 * DIM + d]  = (_Float16)agg[(size_t)b * LSEQ * DIM + d]; // agg[:,0,:]
    vb[3 * DIM + d]  = (_Float16)0.0f;                            // h0 = 0
    c[(size_t)b * DIM + d]    = 0.0f;
    ctgt[(size_t)b * DIM + d] = 0.0f;
    _Float16* fb = feat16 + (size_t)b * KP_F;
    fb[d]           = (_Float16)se;     // feat col 0..199 = ent[s]
    fb[DIM + d]     = (_Float16)0.0f;   // hW slot (filled later)
    fb[2 * DIM + d] = (_Float16)re;     // feat col 400..599 = rel[r]
    h16[(size_t)b * KP_D + d]    = (_Float16)0.0f;
    int16_[(size_t)b * KP_D + d] = (_Float16)0.0f;
  } else if (d < KP_D) {
    h16[(size_t)b * KP_D + d]    = (_Float16)0.0f;   // K pad of h
    int16_[(size_t)b * KP_D + d] = (_Float16)0.0f;   // K pad of inten_raw
    if (d - DIM < KP_F - 3 * DIM)                    // feat K pad cols 600..607
      feat16[(size_t)b * KP_F + 3 * DIM + (d - DIM)] = (_Float16)0.0f;
  }
}

// ---------------------------------------------------------------------------
// Hawkes-LSTM pointwise step: gates -> (h, c, c_tgt); stage next-step inputs
// ---------------------------------------------------------------------------
__global__ void ghnn_lstm_step(const float* __restrict__ g,
                               float* __restrict__ c, float* __restrict__ ctgt,
                               _Float16* __restrict__ v16, _Float16* __restrict__ h16,
                               const float* __restrict__ agg, const float* __restrict__ dt,
                               int t)
{
  int idx = blockIdx.x * blockDim.x + threadIdx.x;
  if (idx >= BATCH * DIM) return;
  int b = idx / DIM, d = idx - b * DIM;
  const float* gr = g + (size_t)b * NP_G;
  float ig  = sigmoidf_(gr[d]);
  float fg  = sigmoidf_(gr[DIM + d]);
  float og  = sigmoidf_(gr[2 * DIM + d]);
  float itg = sigmoidf_(gr[3 * DIM + d]);
  float ftg = sigmoidf_(gr[4 * DIM + d]);
  float z   = tanhf(gr[5 * DIM + d]);
  float dec = softplus_b_(gr[6 * DIM + d]);
  float ci = fg * c[idx] + ig * z;
  float ct = ftg * ctgt[idx] + itg * z;
  float dtb = dt[(size_t)b * LSEQ + t];
  float cdec = ct + (ci - ct) * __expf(-dec * dtb);
  float h = og * tanhf(cdec);
  c[idx] = cdec;
  ctgt[idx] = ct;
  v16[(size_t)b * KP_V + 3 * DIM + d] = (_Float16)h;   // h part of next v
  h16[(size_t)b * KP_D + d]           = (_Float16)h;   // A operand of h@W_h
  if (t + 1 < LSEQ)
    v16[(size_t)b * KP_V + 2 * DIM + d] =
        (_Float16)agg[((size_t)b * LSEQ + (t + 1)) * DIM + d];
}

// hW (f32 [BATCH,DIM]) -> feat16 middle columns
__global__ void ghnn_fill_feat(const float* __restrict__ hW, _Float16* __restrict__ feat16)
{
  int idx = blockIdx.x * blockDim.x + threadIdx.x;
  if (idx >= BATCH * DIM) return;
  int b = idx / DIM, n = idx - b * DIM;
  feat16[(size_t)b * KP_F + DIM + n] = (_Float16)hW[idx];
}

// ---------------------------------------------------------------------------
extern "C" void kernel_launch(void* const* d_in, const int* in_sizes, int n_in,
                              void* d_out, int out_size, void* d_ws, size_t ws_size,
                              hipStream_t stream)
{
  const float* dt   = (const float*)d_in[0];
  const int*   s    = (const int*)d_in[1];
  const int*   r    = (const int*)d_in[2];
  // d_in[3] = o (unused by reference output)
  const int*   hev  = (const int*)d_in[4];
  const int*   hlen = (const int*)d_in[5];
  const float* ent  = (const float*)d_in[6];
  const float* rel  = (const float*)d_in[7];
  const float* Wg   = (const float*)d_in[8];   // [800, 1400]
  const float* bg   = (const float*)d_in[9];   // [1400]
  const float* Wh   = (const float*)d_in[10];  // [200, 200]
  const float* Wi   = (const float*)d_in[11];  // [600, 200]
  float* out = (float*)d_out;
  (void)in_sizes; (void)n_in; (void)out_size; (void)ws_size;

  char* ws = (char*)d_ws;
  size_t off = 0;
  auto take = [&](size_t bytes) -> char* {
    char* p = ws + off;
    off = (off + bytes + 255) & ~(size_t)255;
    return p;
  };
  float*    agg    = (float*)   take((size_t)BATCH * LSEQ * DIM * 4); // 16.4 MB
  _Float16* v16    = (_Float16*)take((size_t)BATCH * KP_V * 2);       //  3.3 MB
  _Float16* WgT    = (_Float16*)take((size_t)NP_G * KP_V * 2);        //  2.25 MB
  float*    g      = (float*)   take((size_t)BATCH * NP_G * 4);       // 11.5 MB
  float*    c      = (float*)   take((size_t)BATCH * DIM * 4);
  float*    ctgt   = (float*)   take((size_t)BATCH * DIM * 4);
  _Float16* h16    = (_Float16*)take((size_t)BATCH * KP_D * 2);
  _Float16* WhT    = (_Float16*)take((size_t)NP_D * KP_D * 2);
  float*    hW     = (float*)   take((size_t)BATCH * DIM * 4);
  _Float16* feat16 = (_Float16*)take((size_t)BATCH * KP_F * 2);
  _Float16* WiT    = (_Float16*)take((size_t)NP_D * KP_F * 2);
  _Float16* int16_ = (_Float16*)take((size_t)BATCH * KP_D * 2);
  _Float16* e16    = (_Float16*)take((size_t)NP_E * KP_D * 2);        //  9.0 MB

  // 1) aggregator (L2-resident gathers)
  ghnn_aggregate<<<dim3(BATCH * LSEQ), dim3(256), 0, stream>>>(hev, hlen, ent, agg);

  // 2) weight prep (transpose + f16 convert, zero padded) — cheap, per call
  {
    int n = NP_G * KP_V;
    ghnn_prep_wT<<<dim3((n + 255) / 256), dim3(256), 0, stream>>>(Wg, WgT, 4 * DIM, 7 * DIM, NP_G, KP_V);
  }
  {
    int n = NP_D * KP_D;
    ghnn_prep_wT<<<dim3((n + 255) / 256), dim3(256), 0, stream>>>(Wh, WhT, DIM, DIM, NP_D, KP_D);
  }
  {
    int n = NP_D * KP_F;
    ghnn_prep_wT<<<dim3((n + 255) / 256), dim3(256), 0, stream>>>(Wi, WiT, 3 * DIM, DIM, NP_D, KP_F);
  }
  {
    int n = NP_E * KP_D;
    ghnn_conv_ent<<<dim3((n + 255) / 256), dim3(256), 0, stream>>>(ent, e16);
  }

  // 3) per-call state init
  ghnn_setup<<<dim3(BATCH), dim3(256), 0, stream>>>(s, r, ent, rel, agg, v16, h16, c, ctgt,
                                                    feat16, int16_);

  // grid: x = N/64 tiles, y = M/128 row-blocks (8 waves * 16 rows)
  auto gemm_grid = [](int M, int Npad) { return dim3(Npad / 64, M / 128); };

  // 4) Hawkes LSTM: 10 dependent [2048,800]x[800,1408] WMMA GEMMs + pointwise
  const int ew_blocks = (BATCH * DIM + 255) / 256;
  for (int t = 0; t < LSEQ; ++t) {
    ghnn_wmma_gemm<KP_V><<<gemm_grid(BATCH, NP_G), dim3(256), 0, stream>>>(
        v16, WgT, g, (_Float16*)nullptr, bg,
        7 * DIM, NP_G, 0, /*mode=*/0);
    ghnn_lstm_step<<<dim3(ew_blocks), dim3(256), 0, stream>>>(g, c, ctgt, v16, h16, agg, dt, t);
  }

  // 5) hW = h @ W_h   [2048,224]x[224,256] -> f32 [2048,200]
  ghnn_wmma_gemm<KP_D><<<gemm_grid(BATCH, NP_D), dim3(256), 0, stream>>>(
      h16, WhT, hW, (_Float16*)nullptr, (const float*)nullptr,
      DIM, DIM, 0, /*mode=*/0);
  ghnn_fill_feat<<<dim3(ew_blocks), dim3(256), 0, stream>>>(hW, feat16);

  // 6) inten_raw = feat @ W_inten   [2048,608]x[608,256] -> f16 [2048,224] (padded)
  ghnn_wmma_gemm<KP_F><<<gemm_grid(BATCH, NP_D), dim3(256), 0, stream>>>(
      feat16, WiT, (float*)nullptr, int16_, (const float*)nullptr,
      DIM, 0, KP_D, /*mode=*/2);

  // 7) intens = softplus_b(inten_raw @ ent^T)  [2048,224]x[224,20032] -> d_out [2048,20000]
  //    164 MB f32 store -> ~7 us HBM floor; B tiles TDM-staged (double buffered).
  ghnn_wmma_gemm<KP_D><<<gemm_grid(BATCH, NP_E), dim3(256), 0, stream>>>(
      int16_, e16, out, (_Float16*)nullptr, (const float*)nullptr,
      NE, NE, 0, /*mode=*/1);
}
